// GPM_34514357190935
// MI455X (gfx1250) — matrix operand order repaired and checked by reading, compiled, not run on hardware
//
#include <hip/hip_runtime.h>
#include <hip/hip_bf16.h>

// Problem constants (match reference)
#define B_    128
#define N_    1000
#define T_    50
#define E_    32000
#define R_    4
#define NSEL_ 500
#define FS_   43            // 2*CF + F0 = 43
#define M_    (B_ * N_)     // 128000 rows for the relation GEMM
#define KP    64            // K padded 43 -> 64 (two K=32 WMMA steps)
#define NP    240           // 5 groups (4 relations + root) * 48 padded cols
#define XS    48            // padded per-node feature stride (f32 buffers)

typedef __attribute__((ext_vector_type(16))) __bf16 v16bf;
typedef __attribute__((ext_vector_type(8)))  __bf16 v8bf;
typedef __attribute__((ext_vector_type(8)))  float  v8f;

// ---------------------------------------------------------------- utilities
__global__ void fill_f32(float* __restrict__ p, int n) {
    int i = blockIdx.x * blockDim.x + threadIdx.x;
    if (i < n) p[i] = 0.0f;
}
__global__ void fill_u32(unsigned* __restrict__ p, int n) {
    int i = blockIdx.x * blockDim.x + threadIdx.x;
    if (i < n) p[i] = 0u;
}

// --------------------------------------------------- transpose/pack weights
// Wt[n][k] column-major (K contiguous per output column), bf16, zero padded.
// n in [0,240): groups of 48; groups 0..3 = w_rel[r], group 4 = w_root.
__global__ void prep_wt(const float* __restrict__ w_rel,
                        const float* __restrict__ w_root,
                        __hip_bfloat16* __restrict__ wt) {
    int i = blockIdx.x * blockDim.x + threadIdx.x;
    if (i >= NP * KP) return;
    int n = i / KP, k = i % KP;
    float v = 0.0f;
    if (k < FS_) {
        int grp = n / 48, g = n % 48;
        if (g < FS_) {
            if (grp < 4) v = w_rel[((size_t)grp * FS_ + k) * FS_ + g];
            else         v = w_root[(size_t)k * FS_ + g];
        }
    }
    wt[(size_t)n * KP + k] = __float2bfloat16(v);
}

// ------------------------------------------------- stage 1: temporal convs
// One thread per (b,n) slot. Weights broadcast from LDS; obs rows live in
// LDS with stride 151 (odd) so 64 banks never conflict across the wave.
__global__ __launch_bounds__(64) void temporal_kernel(
    const float* __restrict__ obs,
    const float* __restrict__ ws1, const float* __restrict__ bs1,
    const float* __restrict__ ws2, const float* __restrict__ bs2,
    const float* __restrict__ wm1, const float* __restrict__ bm1,
    const float* __restrict__ wm2, const float* __restrict__ bm2,
    float* __restrict__ xf32, __hip_bfloat16* __restrict__ xbf) {
    __shared__ float s_ws1[27], s_bs1[3], s_ws2[2880], s_bs2[20];
    __shared__ float s_wm1[189], s_bm1[3], s_wm2[1800], s_bm2[20];
    __shared__ float s_obs[64 * 151];

    int tid = threadIdx.x;
    for (int i = tid; i < 27;   i += 64) s_ws1[i] = ws1[i];
    for (int i = tid; i < 3;    i += 64) { s_bs1[i] = bs1[i]; s_bm1[i] = bm1[i]; }
    for (int i = tid; i < 2880; i += 64) s_ws2[i] = ws2[i];
    for (int i = tid; i < 20;   i += 64) { s_bs2[i] = bs2[i]; s_bm2[i] = bm2[i]; }
    for (int i = tid; i < 189;  i += 64) s_wm1[i] = wm1[i];
    for (int i = tid; i < 1800; i += 64) s_wm2[i] = wm2[i];

    int gid = blockIdx.x * 64 + tid;          // exact: grid = M_/64
    int b = gid / N_, n = gid % N_;
    float* ob = &s_obs[tid * 151];
    for (int c = 0; c < 3; ++c)
        for (int t = 0; t < T_; ++t)
            ob[c * T_ + t] = obs[(((size_t)b * 3 + c) * N_ + n) * T_ + t];
    __syncthreads();

    // short branch: conv(k=3) -> relu -> conv(k=48) -> relu
    float accS[20];
    for (int o = 0; o < 20; ++o) accS[o] = s_bs2[o];
    for (int k = 0; k < 48; ++k) {
        for (int i = 0; i < 3; ++i) {
            float s = s_bs1[i];
            for (int c = 0; c < 3; ++c)
                for (int kw = 0; kw < 3; ++kw)
                    s += ob[c * T_ + k + kw] * s_ws1[(i * 3 + c) * 3 + kw];
            s = s > 0.0f ? s : 0.0f;
            for (int o = 0; o < 20; ++o) accS[o] += s * s_ws2[(o * 3 + i) * 48 + k];
        }
    }
    // mid branch: conv(k=21) -> relu -> conv(k=30) -> relu
    float accM[20];
    for (int o = 0; o < 20; ++o) accM[o] = s_bm2[o];
    for (int k = 0; k < 30; ++k) {
        for (int i = 0; i < 3; ++i) {
            float s = s_bm1[i];
            for (int c = 0; c < 3; ++c)
                for (int kw = 0; kw < 21; ++kw)
                    s += ob[c * T_ + k + kw] * s_wm1[(i * 3 + c) * 21 + kw];
            s = s > 0.0f ? s : 0.0f;
            for (int o = 0; o < 20; ++o) accM[o] += s * s_wm2[(o * 3 + i) * 30 + k];
        }
    }
    // long branch: relu(max over T)
    float lg[3];
    for (int c = 0; c < 3; ++c) {
        float mx = ob[c * T_];
        for (int t = 1; t < T_; ++t) mx = fmaxf(mx, ob[c * T_ + t]);
        lg[c] = mx > 0.0f ? mx : 0.0f;
    }

    float* xo = xf32 + (size_t)gid * XS;
    __hip_bfloat16* xb = xbf + (size_t)gid * KP;
    for (int o = 0; o < 20; ++o) {
        float vs = accS[o] > 0.0f ? accS[o] : 0.0f;
        float vm = accM[o] > 0.0f ? accM[o] : 0.0f;
        xo[o] = vs;      xb[o]      = __float2bfloat16(vs);
        xo[20 + o] = vm; xb[20 + o] = __float2bfloat16(vm);
    }
    for (int c = 0; c < 3; ++c) { xo[40 + c] = lg[c]; xb[40 + c] = __float2bfloat16(lg[c]); }
    for (int f = FS_; f < XS; ++f) xo[f] = 0.0f;
    for (int f = FS_; f < KP; ++f) xb[f] = __float2bfloat16(0.0f);
}

// -------------------------------------- stage 2: relation GEMM via WMMA
// C(M x 240) = A(M x 64, bf16) * Wt^T.
// One wave per 16-row strip: A fragments (both K-steps) are loaded ONCE into
// registers and reused across all 15 N-tiles (B is 30 KB, L1/L2 resident),
// cutting A read traffic 15x vs. one-tile-per-wave. 30 WMMAs per wave.
__global__ __launch_bounds__(32) void gemm_xw(
    const __hip_bfloat16* __restrict__ xbf,
    const __hip_bfloat16* __restrict__ wt,
    float* __restrict__ xw) {
    int lane = threadIdx.x;
    int m0 = blockIdx.x * 16;
    int hi = lane >> 4;      // half-wave select
    int lo = lane & 15;

    // A fragment (16-bit A 16x32 layout): halves 0-7 = K[k0..k0+8),
    // halves 8-15 = K[k0+16..k0+24), k0 = kk + hi*8 -> two b128 loads each.
    const __hip_bfloat16* arow = xbf + (size_t)(m0 + lo) * KP;
    union { v16bf v; v8bf h[2]; } A0, A1;
    {
        int ka = hi * 8;
        A0.h[0] = *(const v8bf*)(arow + ka);
        A0.h[1] = *(const v8bf*)(arow + ka + 16);
        A1.h[0] = *(const v8bf*)(arow + 32 + ka);
        A1.h[1] = *(const v8bf*)(arow + 32 + ka + 16);
    }
    for (int nt = 0; nt < NP / 16; ++nt) {
        int n0 = nt * 16;
        // B fragment: same K order as A: halves 0-7 = K[k0..k0+8),
        // halves 8-15 = K[k0+16..k0+24), k0 = kk + hi*8; two b128 loads each.
        const __hip_bfloat16* bcol = wt + (size_t)(n0 + lo) * KP;
        union { v16bf v; v8bf h[2]; } B0, B1;
        int kb = hi * 8;
        B0.h[0] = *(const v8bf*)(bcol + kb);
        B0.h[1] = *(const v8bf*)(bcol + kb + 16);
        B1.h[0] = *(const v8bf*)(bcol + 32 + kb);
        B1.h[1] = *(const v8bf*)(bcol + 32 + kb + 16);
        v8f c = {};
        c = __builtin_amdgcn_wmma_f32_16x16x32_bf16(
                false, A0.v, false, B0.v, (short)0, c, false, false);
        c = __builtin_amdgcn_wmma_f32_16x16x32_bf16(
                false, A1.v, false, B1.v, (short)0, c, false, false);
        // C layout: col = lane&15, row = vgpr + (lane>>4)*8
        float* out = xw + (size_t)(m0 + hi * 8) * NP + (n0 + lo);
        #pragma unroll
        for (int v = 0; v < 8; ++v) out[(size_t)v * NP] = c[v];
    }
}

// ---------------------------------------------- stage 3: edge aggregation
__global__ void edge_count(const int* __restrict__ ei, const int* __restrict__ et,
                           unsigned* __restrict__ cnt) {
    int e = blockIdx.x * blockDim.x + threadIdx.x;
    if (e < E_) atomicAdd(&cnt[ei[E_ + e] * R_ + et[e]], 1u);
}
__global__ void seg_inv(const unsigned* __restrict__ cnt, float* __restrict__ inv) {
    int s = blockIdx.x * blockDim.x + threadIdx.x;
    if (s < N_ * R_) inv[s] = cnt[s] ? 1.0f / (float)cnt[s] : 0.0f;
}
// agg[b,dst,f] += inv[dst*R+r] * xw[b,src,r*48+f]  (== segment-mean then sum-r)
__global__ __launch_bounds__(64) void scatter_msgs(
    const float* __restrict__ xw, const int* __restrict__ ei,
    const int* __restrict__ et, const float* __restrict__ inv,
    float* __restrict__ agg) {
    int e = blockIdx.x;
    int b = blockIdx.y;
    int f = threadIdx.x;
    int src = ei[e], dst = ei[E_ + e], r = et[e];
    float w = inv[dst * R_ + r];
    if (f < FS_) {
        float v = xw[((size_t)b * N_ + src) * NP + r * 48 + f] * w;
        atomicAdd(&agg[((size_t)b * N_ + dst) * XS + f], v);
    }
}

// ---------------------------------- stage 4: gout = leaky_relu(agg+root+b)
__global__ void gout_kernel(const float* __restrict__ agg,
                            const float* __restrict__ xw,
                            const float* __restrict__ bg,
                            float* __restrict__ gout) {
    int i = blockIdx.x * blockDim.x + threadIdx.x;
    if (i >= M_ * XS) return;
    int m = i / XS, f = i % XS;
    float v = 0.0f;
    if (f < FS_) {
        v = agg[i] + xw[(size_t)m * NP + 192 + f] + bg[f];
        v = v > 0.0f ? v : 0.01f * v;
    }
    gout[i] = v;
}

// ------------------------------------- stage 5: final head + softmax(501)
__global__ __launch_bounds__(512) void final_softmax(
    const float* __restrict__ xf32, const float* __restrict__ gout,
    const float* __restrict__ la,   const float* __restrict__ wfin,
    const float* __restrict__ bfin, const int* __restrict__ ns,
    float* __restrict__ out) {
    __shared__ float s_red[512];
    __shared__ float s_w[87];
    int b = blockIdx.x, tid = threadIdx.x;
    if (tid < 87) s_w[tid] = wfin[tid];
    __syncthreads();

    bool valid = tid <= NSEL_;
    float logit = -3.4e38f;
    if (tid == 0) {
        logit = 0.0f;                                   // prepended zero logit
    } else if (valid) {
        int node = ns[tid - 1];
        const float* tp = xf32 + ((size_t)b * N_ + node) * XS;
        const float* gp = gout + ((size_t)b * N_ + node) * XS;
        float acc = bfin[0] + s_w[0] * la[(size_t)b * (NSEL_ + 1) + tid];
        for (int f = 0; f < FS_; ++f)
            acc += s_w[1 + f] * tp[f] + s_w[44 + f] * gp[f];
        logit = acc;
    }
    s_red[tid] = logit;
    __syncthreads();
    for (int o = 256; o > 0; o >>= 1) {
        if (tid < o) s_red[tid] = fmaxf(s_red[tid], s_red[tid + o]);
        __syncthreads();
    }
    float mx = s_red[0];
    __syncthreads();
    float ev = valid ? __expf(logit - mx) : 0.0f;
    s_red[tid] = ev;
    __syncthreads();
    for (int o = 256; o > 0; o >>= 1) {
        if (tid < o) s_red[tid] += s_red[tid + o];
        __syncthreads();
    }
    float sum = s_red[0];
    if (valid) out[(size_t)b * (NSEL_ + 1) + tid] = ev / sum;
}

// ------------------------------------------------------------------ driver
extern "C" void kernel_launch(void* const* d_in, const int* in_sizes, int n_in,
                              void* d_out, int out_size, void* d_ws, size_t ws_size,
                              hipStream_t stream) {
    const float* obs    = (const float*)d_in[0];
    const float* la     = (const float*)d_in[1];
    const float* ws1    = (const float*)d_in[2];
    const float* bs1    = (const float*)d_in[3];
    const float* ws2    = (const float*)d_in[4];
    const float* bs2    = (const float*)d_in[5];
    const float* wm1    = (const float*)d_in[6];
    const float* bm1    = (const float*)d_in[7];
    const float* wm2    = (const float*)d_in[8];
    const float* bm2    = (const float*)d_in[9];
    const float* w_rel  = (const float*)d_in[10];
    const float* w_root = (const float*)d_in[11];
    const float* b_g    = (const float*)d_in[12];
    const float* w_fin  = (const float*)d_in[13];
    const float* b_fin  = (const float*)d_in[14];
    const int*   ei     = (const int*)d_in[15];
    const int*   et     = (const int*)d_in[16];
    const int*   ns     = (const int*)d_in[17];
    float* out = (float*)d_out;

    char* ws = (char*)d_ws;
    size_t off = 0;
    auto wsalloc = [&](size_t bytes) -> void* {
        void* p = ws + off;
        off = (off + bytes + 255) & ~(size_t)255;
        return p;
    };
    float*          xf32 = (float*)wsalloc((size_t)M_ * XS * sizeof(float));
    __hip_bfloat16* xbf  = (__hip_bfloat16*)wsalloc((size_t)M_ * KP * 2);
    __hip_bfloat16* wt   = (__hip_bfloat16*)wsalloc((size_t)NP * KP * 2);
    float*          xw   = (float*)wsalloc((size_t)M_ * NP * sizeof(float));
    float*          agg  = (float*)wsalloc((size_t)M_ * XS * sizeof(float));
    float*          gbuf = (float*)wsalloc((size_t)M_ * XS * sizeof(float));
    unsigned*       cnt  = (unsigned*)wsalloc((size_t)N_ * R_ * sizeof(unsigned));
    float*          inv  = (float*)wsalloc((size_t)N_ * R_ * sizeof(float));

    fill_f32<<<(M_ * XS + 255) / 256, 256, 0, stream>>>(agg, M_ * XS);
    fill_u32<<<(N_ * R_ + 255) / 256, 256, 0, stream>>>(cnt, N_ * R_);
    prep_wt<<<(NP * KP + 255) / 256, 256, 0, stream>>>(w_rel, w_root, wt);

    temporal_kernel<<<M_ / 64, 64, 0, stream>>>(
        obs, ws1, bs1, ws2, bs2, wm1, bm1, wm2, bm2, xf32, xbf);

    gemm_xw<<<M_ / 16, 32, 0, stream>>>(xbf, wt, xw);

    edge_count<<<(E_ + 255) / 256, 256, 0, stream>>>(ei, et, cnt);
    seg_inv<<<(N_ * R_ + 255) / 256, 256, 0, stream>>>(cnt, inv);
    scatter_msgs<<<dim3(E_, B_), 64, 0, stream>>>(xw, ei, et, inv, agg);

    gout_kernel<<<(M_ * XS + 255) / 256, 256, 0, stream>>>(agg, xw, b_g, gbuf);

    final_softmax<<<B_, 512, 0, stream>>>(xf32, gbuf, la, w_fin, b_fin, ns, out);
}
